// VarianceAdaptor_61168924229710
// MI455X (gfx1250) — matrix operand order repaired and checked by reading, not compiled, not run on hardware
//
#include <hip/hip_runtime.h>
#include <stdint.h>

// Problem dims
#define Bn   64
#define Tn   2000
#define Fn   256
#define ROWS (Bn * Tn)          // 128000 flattened (b,t) rows
#define ZON  0.1f
#define NWG  16                 // persistent workgroups in the sequential scan

typedef __attribute__((ext_vector_type(16))) __bf16 v16bf;
typedef __attribute__((ext_vector_type(8)))  float  v8f;

union FragU { uint4 u[2]; v16bf v; };

__device__ __forceinline__ unsigned short f2bf(float f) {
  unsigned u = __float_as_uint(f);
  u += 0x7fffu + ((u >> 16) & 1u);          // round-to-nearest-even
  return (unsigned short)(u >> 16);
}
__device__ __forceinline__ float bf2f(unsigned short h) {
  return __uint_as_float(((unsigned)h) << 16);
}
__device__ __forceinline__ float sigmoidf(float x) { return 1.0f / (1.0f + __expf(-x)); }
__device__ __forceinline__ v8f vzero8() { v8f z = {0,0,0,0,0,0,0,0}; return z; }

// A fragment (16x32 bf16, MxK): lane m=l&15; lanes0-15 take K = k0+{0..7} and k0+16+{0..7},
// lanes16-31 take K = k0+8+{0..7} and k0+24+{0..7}  (ISA 7.12.2 16-bit A table)
__device__ __forceinline__ v16bf load_a(const unsigned short* __restrict__ base,
                                        long row, int ldk, int k0, int lane, bool valid) {
  FragU f;
  if (valid) {
    int sel8 = (lane & 16) ? 8 : 0;
    const unsigned short* p = base + (size_t)row * ldk + k0 + sel8;
    f.u[0] = *(const uint4*)(p);
    f.u[1] = *(const uint4*)(p + 16);
  } else {
    f.u[0] = make_uint4(0, 0, 0, 0);
    f.u[1] = make_uint4(0, 0, 0, 0);
  }
  return f.v;
}

// B fragment (32x16 bf16, KxN), B[k][n] = W[n][k] for row-major W[N][K]:
// lane n=l&15; lanes0-15 hold K=k0+{0..7} and k0+16+{0..7}, lanes16-31 hold K=k0+8+{0..7} and k0+24+{0..7}
__device__ __forceinline__ v16bf load_b(const unsigned short* __restrict__ W,
                                        int nrow, int ldk, int k0, int lane) {
  FragU f;
  int sel16 = (lane & 16) ? 8 : 0;
  const unsigned short* p = W + (size_t)nrow * ldk + k0 + sel16;
  f.u[0] = *(const uint4*)(p);
  f.u[1] = *(const uint4*)(p + 16);
  return f.v;
}

__device__ __forceinline__ v8f wmma_bf16(v16bf a, v16bf b, v8f c) {
  return __builtin_amdgcn_wmma_f32_16x16x32_bf16(false, a, false, b, (short)0, c, false, false);
}

// ---------------------------------------------------------------- prep kernels

__global__ void k_cvt_bf16(const float* __restrict__ src, unsigned short* __restrict__ dst, long n) {
  long i = (long)blockIdx.x * blockDim.x + threadIdx.x;
  long stride = (long)gridDim.x * blockDim.x;
  for (; i < n; i += stride) dst[i] = f2bf(src[i]);
}

// conv w[tap][cin][cout] f32 -> ws[tap][cout][cin] bf16 (so B columns become contiguous rows)
__global__ void k_cvt_convw(const float* __restrict__ w, unsigned short* __restrict__ ws) {
  long n = 3L * 256 * 256;
  long i = (long)blockIdx.x * blockDim.x + threadIdx.x;
  long stride = (long)gridDim.x * blockDim.x;
  for (; i < n; i += stride) {
    int tap = (int)(i / 65536);
    int r   = (int)(i % 65536);
    int cout = r / 256, cin = r % 256;
    ws[i] = f2bf(w[(size_t)tap * 65536 + (size_t)cin * 256 + cout]);
  }
}

// rank-1 prenet folding: u[n] = Wp[n,:]·prenet_w, v[n] = Wp[n,:]·prenet_b + lstm0_b[n]
__global__ void k_uv(const float* __restrict__ wih0, const float* __restrict__ pw,
                     const float* __restrict__ pb, const float* __restrict__ b0,
                     float* __restrict__ u, float* __restrict__ v) {
  int n = blockIdx.x * blockDim.x + threadIdx.x;
  if (n >= 1024) return;
  const float* rowp = wih0 + (size_t)n * 512 + 256;
  float su = 0.f, sv = 0.f;
  for (int k = 0; k < 256; ++k) { su += rowp[k] * pw[k]; sv += rowp[k] * pb[k]; }
  u[n] = su; v[n] = sv + b0[n];
}

// ------------------------------------------------- conv(K=3) + ReLU + LayerNorm
// one WG = 16 (b,t) rows x full 256 channels; 8 waves x 2 N-tiles; 3 taps x 8 K-tiles
__global__ __launch_bounds__(256) void k_conv_ln(
    const unsigned short* __restrict__ xin,   // [ROWS][256] bf16
    const unsigned short* __restrict__ wsw,   // [3][256 cout][256 cin] bf16
    const float* __restrict__ bias,
    const float* __restrict__ gamma, const float* __restrict__ beta,
    unsigned short* __restrict__ yout)        // [ROWS][256] bf16
{
  __shared__ float ysh[16][256];
  __shared__ float lmean[16], lrstd[16];
  int tid = threadIdx.x, lane = tid & 31, wid = tid >> 5;
  int rowbase = blockIdx.x * 16;              // 2000 % 16 == 0 -> never crosses batch
  int b = rowbase / Tn, tbase = rowbase % Tn;
  int m = lane & 15;
  int sel8 = (lane & 16) ? 8 : 0;
  int n0 = wid * 32, n1 = n0 + 16;

  v8f acc0 = vzero8(), acc1 = vzero8();
  for (int tap = 0; tap < 3; ++tap) {
    int tt = tbase + m + tap - 1;
    bool valid = (tt >= 0) && (tt < Tn);
    long arow = (long)b * Tn + tt;
    const unsigned short* wt = wsw + (size_t)tap * 65536;
    for (int kt = 0; kt < 8; ++kt) {
      int k0 = kt * 32;
      v16bf a  = load_a(xin, arow, 256, k0, lane, valid);
      v16bf b0 = load_b(wt, n0 + m, 256, k0, lane);
      v16bf b1 = load_b(wt, n1 + m, 256, k0, lane);
      acc0 = wmma_bf16(a, b0, acc0);
      acc1 = wmma_bf16(a, b1, acc1);
    }
  }
  float bia0 = bias[n0 + m], bia1 = bias[n1 + m];
#pragma unroll
  for (int i = 0; i < 8; ++i) {
    float x0 = fmaxf(acc0[i] + bia0, 0.0f);
    float x1 = fmaxf(acc1[i] + bia1, 0.0f);
    acc0[i] = x0; acc1[i] = x1;
    ysh[sel8 + i][n0 + m] = x0;
    ysh[sel8 + i][n1 + m] = x1;
  }
  __syncthreads();
  { // LayerNorm stats: 16 threads per row, shfl-reduce within 16-lane groups
    int r = tid >> 4, j = tid & 15;
    float s = 0.f, s2 = 0.f;
    for (int c = j; c < 256; c += 16) { float v = ysh[r][c]; s += v; s2 += v * v; }
    for (int o = 8; o > 0; o >>= 1) { s += __shfl_xor(s, o, 16); s2 += __shfl_xor(s2, o, 16); }
    if (j == 0) {
      float mu = s * (1.0f / 256.0f);
      float var = s2 * (1.0f / 256.0f) - mu * mu;
      lmean[r] = mu; lrstd[r] = rsqrtf(var + 1e-5f);
    }
  }
  __syncthreads();
  float g0 = gamma[n0 + m], g1 = gamma[n1 + m];
  float be0 = beta[n0 + m], be1 = beta[n1 + m];
#pragma unroll
  for (int i = 0; i < 8; ++i) {
    int r = sel8 + i;
    long orow = (long)rowbase + r;
    float mu = lmean[r], rs = lrstd[r];
    yout[orow * 256 + n0 + m] = f2bf((acc0[i] - mu) * rs * g0 + be0);
    yout[orow * 256 + n1 + m] = f2bf((acc1[i] - mu) * rs * g1 + be1);
  }
}

// ------------- hoisted input GEMM: gates_pre[r][n] = enc[r]·wih0[n,0:256] + prev*u[n] + v[n]
__global__ __launch_bounds__(256) void k_gates_pre(
    const unsigned short* __restrict__ enc,   // [ROWS][256] bf16
    const unsigned short* __restrict__ wih0b, // [1024][512] bf16 (cols 0..255 used)
    const float* __restrict__ target,         // [B][T]
    const float* __restrict__ uvec, const float* __restrict__ vvec,
    float* __restrict__ gp)                   // [ROWS][1024] f32
{
  int tid = threadIdx.x, lane = tid & 31, wid = tid >> 5;
  int rowbase = blockIdx.x * 16;
  int m = lane & 15;
  int sel8 = (lane & 16) ? 8 : 0;
  int n = blockIdx.y * 128 + wid * 16 + m;
  long arow = (long)rowbase + m;

  v8f acc = vzero8();
  for (int kt = 0; kt < 8; ++kt) {
    int k0 = kt * 32;
    v16bf a  = load_a(enc, arow, 256, k0, lane, true);
    v16bf bf = load_b(wih0b, n, 512, k0, lane);
    acc = wmma_bf16(a, bf, acc);
  }
  float un = uvec[n], vn = vvec[n];
  int b = rowbase / Tn, tb = rowbase % Tn;
#pragma unroll
  for (int i = 0; i < 8; ++i) {
    int r = sel8 + i;
    int t = tb + r;
    float prev = (t == 0) ? 0.0f : target[(size_t)b * Tn + t - 1];
    gp[((size_t)rowbase + r) * 1024 + n] = acc[i] + prev * un + vn;
  }
}

// out_base[b,t] = enc[b,t,:]·out_w[256:512] + out_b  (one wave per row)
__global__ __launch_bounds__(256) void k_out_base(
    const unsigned short* __restrict__ enc, const float* __restrict__ outw,
    const float* __restrict__ outb, float* __restrict__ out)
{
  int wid = threadIdx.x >> 5, lane = threadIdx.x & 31;
  long row = (long)blockIdx.x * 8 + wid;
  if (row >= ROWS) return;
  const unsigned short* p = enc + row * 256 + lane * 8;
  float s = 0.f;
#pragma unroll
  for (int e = 0; e < 8; ++e) s += bf2f(p[e]) * outw[256 + lane * 8 + e];
  for (int o = 16; o > 0; o >>= 1) s += __shfl_xor(s, o, 32);
  if (lane == 0) out[row] = s + outb[0];
}

__global__ void k_mask(const unsigned char* __restrict__ mask, float* __restrict__ out, int n) {
  int i = blockIdx.x * blockDim.x + threadIdx.x;
  if (i < n && mask[i]) out[i] = 0.0f;
}

// ---------------------------------------------------------- sequential LSTM scan
__device__ __forceinline__ void grid_barrier(unsigned* bar, unsigned& cnt) {
  __syncthreads();
  __threadfence();                                          // release my WG's writes
  ++cnt;
  if (threadIdx.x == 0) {
    __hip_atomic_fetch_add(bar, 1u, __ATOMIC_RELEASE, __HIP_MEMORY_SCOPE_AGENT);
    unsigned target = cnt * NWG;
    while (__hip_atomic_load(bar, __ATOMIC_ACQUIRE, __HIP_MEMORY_SCOPE_AGENT) < target)
      __builtin_amdgcn_s_sleep(2);
  }
  __syncthreads();
  __threadfence();                                          // acquire others' writes
}

__global__ __launch_bounds__(256, 1) void k_seq(
    const float* __restrict__ gp,              // [ROWS][1024]
    const unsigned short* __restrict__ whh0b,  // [1024][256] bf16
    const unsigned short* __restrict__ wih1b,  // [1024][256] bf16
    const unsigned short* __restrict__ whh1b,  // [1024][256] bf16
    const float* __restrict__ b1,              // [1024]
    const float* __restrict__ outw,            // [512]
    unsigned short* __restrict__ h0buf,        // [2][64][256] bf16 (double buffered)
    unsigned short* __restrict__ h1buf,        // [2][64][256] bf16
    float* __restrict__ out,                   // [B][T]
    unsigned* __restrict__ bar)
{
  __shared__ float gbuf[4][64][16];            // raw i/f/g/o pre-activations (this WG's slice)
  __shared__ float c0s[64][16], h0s[64][16], c1s[64][16], h1s[64][16];
  __shared__ float wout[16];

  int tid = threadIdx.x, lane = tid & 31, wid = tid >> 5;
  int wg = blockIdx.x;
  int js = wg * 16;                            // this WG owns hidden columns [js, js+16)
  int m = lane & 15;
  int sel8 = (lane & 16) ? 8 : 0;

  for (int e = tid; e < 64 * 16; e += 256) {
    ((float*)c0s)[e] = 0.f; ((float*)h0s)[e] = 0.f;
    ((float*)c1s)[e] = 0.f; ((float*)h1s)[e] = 0.f;
  }
  if (tid < 16) wout[tid] = outw[js + tid];
  __syncthreads();
  unsigned barcnt = 0;

  // wave -> 2 of 16 (gate, M-tile) tiles
  int ti0 = wid * 2, ti1 = wid * 2 + 1;
  int gi0 = ti0 >> 2, mt0 = ti0 & 3;
  int gi1 = ti1 >> 2, mt1 = ti1 & 3;
  int n_g0 = gi0 * 256 + js + m;
  int n_g1 = gi1 * 256 + js + m;

  for (int t = 0; t < Tn; ++t) {
    int pw = t & 1, pr = pw ^ 1;
    const unsigned short* h0r = h0buf + (size_t)pr * 64 * 256;
    const unsigned short* h1r = h1buf + (size_t)pr * 64 * 256;
    unsigned short* h0w = h0buf + (size_t)pw * 64 * 256;
    unsigned short* h1w = h1buf + (size_t)pw * 64 * 256;

    // ---- LSTM0 recurrent GEMM: h0(t-1) @ whh0^T + gates_pre[t]
    {
      v8f a0 = vzero8(), a1 = vzero8();
      for (int kt = 0; kt < 8; ++kt) {
        int k0 = kt * 32;
        v16bf A0 = load_a(h0r, mt0 * 16 + m, 256, k0, lane, true);
        v16bf B0 = load_b(whh0b, n_g0, 256, k0, lane);
        a0 = wmma_bf16(A0, B0, a0);
        v16bf A1 = load_a(h0r, mt1 * 16 + m, 256, k0, lane, true);
        v16bf B1 = load_b(whh0b, n_g1, 256, k0, lane);
        a1 = wmma_bf16(A1, B1, a1);
      }
#pragma unroll
      for (int i = 0; i < 8; ++i) {
        int r0 = mt0 * 16 + sel8 + i;
        int r1 = mt1 * 16 + sel8 + i;
        a0[i] += gp[((size_t)r0 * Tn + t) * 1024 + n_g0];
        a1[i] += gp[((size_t)r1 * Tn + t) * 1024 + n_g1];
        gbuf[gi0][r0][m] = a0[i];
        gbuf[gi1][r1][m] = a1[i];
      }
      if (t + 1 < Tn)  // stream next step's precomputed gates (global_prefetch_b8)
        __builtin_prefetch(&gp[((size_t)(mt0 * 16 + sel8) * Tn + t + 1) * 1024 + n_g0], 0, 1);
    }
    __syncthreads();
    for (int e = tid; e < 1024; e += 256) {
      int mm = e >> 4, j = e & 15;
      float xi = gbuf[0][mm][j], xf = gbuf[1][mm][j];
      float xg = gbuf[2][mm][j], xo = gbuf[3][mm][j];
      float c = c0s[mm][j], h = h0s[mm][j];
      float craw = sigmoidf(xf) * c + sigmoidf(xi) * tanhf(xg);
      float hraw = sigmoidf(xo) * tanhf(craw);
      float cn = ZON * c + (1.0f - ZON) * craw;
      float hn = ZON * h + (1.0f - ZON) * hraw;
      c0s[mm][j] = cn; h0s[mm][j] = hn;
      h0w[mm * 256 + js + j] = f2bf(hn);
    }
    grid_barrier(bar, barcnt);                 // publish h0(t) to all WGs

    // ---- LSTM1: h0(t) @ wih1^T + h1(t-1) @ whh1^T + b1
    {
      v8f a0 = vzero8(), a1 = vzero8();
      for (int kt = 0; kt < 8; ++kt) {
        int k0 = kt * 32;
        v16bf A0 = load_a(h0w, mt0 * 16 + m, 256, k0, lane, true);
        v16bf B0 = load_b(wih1b, n_g0, 256, k0, lane);
        a0 = wmma_bf16(A0, B0, a0);
        v16bf A1 = load_a(h0w, mt1 * 16 + m, 256, k0, lane, true);
        v16bf B1 = load_b(wih1b, n_g1, 256, k0, lane);
        a1 = wmma_bf16(A1, B1, a1);
      }
      for (int kt = 0; kt < 8; ++kt) {
        int k0 = kt * 32;
        v16bf A0 = load_a(h1r, mt0 * 16 + m, 256, k0, lane, true);
        v16bf B0 = load_b(whh1b, n_g0, 256, k0, lane);
        a0 = wmma_bf16(A0, B0, a0);
        v16bf A1 = load_a(h1r, mt1 * 16 + m, 256, k0, lane, true);
        v16bf B1 = load_b(whh1b, n_g1, 256, k0, lane);
        a1 = wmma_bf16(A1, B1, a1);
      }
      float bb0 = b1[n_g0], bb1 = b1[n_g1];
#pragma unroll
      for (int i = 0; i < 8; ++i) {
        int r0 = mt0 * 16 + sel8 + i;
        int r1 = mt1 * 16 + sel8 + i;
        gbuf[gi0][r0][m] = a0[i] + bb0;
        gbuf[gi1][r1][m] = a1[i] + bb1;
      }
    }
    __syncthreads();
    for (int e = tid; e < 1024; e += 256) {
      int mm = e >> 4, j = e & 15;
      float xi = gbuf[0][mm][j], xf = gbuf[1][mm][j];
      float xg = gbuf[2][mm][j], xo = gbuf[3][mm][j];
      float c = c1s[mm][j], h = h1s[mm][j];
      float craw = sigmoidf(xf) * c + sigmoidf(xi) * tanhf(xg);
      float hraw = sigmoidf(xo) * tanhf(craw);
      float cn = ZON * c + (1.0f - ZON) * craw;
      float hn = ZON * h + (1.0f - ZON) * hraw;
      c1s[mm][j] = cn; h1s[mm][j] = hn;
      h1w[mm * 256 + js + j] = f2bf(hn);
    }
    __syncthreads();
    if (tid < 64) {                            // this slice's output-head contribution
      float s = 0.f;
#pragma unroll
      for (int j = 0; j < 16; ++j) s += h1s[tid][j] * wout[j];
      atomicAdd(&out[(size_t)tid * Tn + t], s);
    }
    grid_barrier(bar, barcnt);                 // publish h1(t)
  }
}

// ---------------------------------------------------------------- host launch

extern "C" void kernel_launch(void* const* d_in, const int* in_sizes, int n_in,
                              void* d_out, int out_size, void* d_ws, size_t ws_size,
                              hipStream_t stream) {
  (void)in_sizes; (void)n_in; (void)out_size; (void)ws_size;
  const float* enc_in  = (const float*)d_in[0];
  const unsigned char* mask = (const unsigned char*)d_in[1];
  const float* target  = (const float*)d_in[2];
  const float* conv0_w = (const float*)d_in[3];
  const float* conv0_b = (const float*)d_in[4];
  const float* ln0_g   = (const float*)d_in[5];
  const float* ln0_b   = (const float*)d_in[6];
  const float* conv1_w = (const float*)d_in[7];
  const float* conv1_b = (const float*)d_in[8];
  const float* ln1_g   = (const float*)d_in[9];
  const float* ln1_b   = (const float*)d_in[10];
  const float* pre_w   = (const float*)d_in[11];
  const float* pre_b   = (const float*)d_in[12];
  const float* l0_wih  = (const float*)d_in[13];
  const float* l0_whh  = (const float*)d_in[14];
  const float* l0_b    = (const float*)d_in[15];
  const float* l1_wih  = (const float*)d_in[16];
  const float* l1_whh  = (const float*)d_in[17];
  const float* l1_b    = (const float*)d_in[18];
  const float* out_w   = (const float*)d_in[19];
  const float* out_b   = (const float*)d_in[20];
  float* out = (float*)d_out;
  char* ws = (char*)d_ws;

  size_t off = 0;
  auto carve = [&](size_t bytes) { size_t o = off; off += (bytes + 255) & ~(size_t)255; return o; };
  const size_t SZ_ACT = (size_t)ROWS * 256 * 2;                 // 65.5 MB bf16 activations
  unsigned short* xbf   = (unsigned short*)(ws + carve(SZ_ACT));
  unsigned short* e1bf  = (unsigned short*)(ws + carve(SZ_ACT));
  unsigned short* e2bf  = (unsigned short*)(ws + carve(SZ_ACT));
  float*          gpre  = (float*)(ws + carve((size_t)ROWS * 1024 * 4));  // 524 MB
  unsigned short* c0sw  = (unsigned short*)(ws + carve(3 * 65536 * 2));
  unsigned short* c1sw  = (unsigned short*)(ws + carve(3 * 65536 * 2));
  unsigned short* wih0b = (unsigned short*)(ws + carve(1024 * 512 * 2));
  unsigned short* whh0b = (unsigned short*)(ws + carve(1024 * 256 * 2));
  unsigned short* wih1b = (unsigned short*)(ws + carve(1024 * 256 * 2));
  unsigned short* whh1b = (unsigned short*)(ws + carve(1024 * 256 * 2));
  float*          uvec  = (float*)(ws + carve(1024 * 4));
  float*          vvec  = (float*)(ws + carve(1024 * 4));
  unsigned short* h0buf = (unsigned short*)(ws + carve(2 * 64 * 256 * 2));
  unsigned short* h1buf = (unsigned short*)(ws + carve(2 * 64 * 256 * 2));
  unsigned*       bar   = (unsigned*)(ws + carve(256));

  // prep: bf16 conversions + conv weight transpose + prenet rank-1 folding
  k_cvt_bf16<<<2048, 256, 0, stream>>>(enc_in, xbf, (long)ROWS * 256);
  k_cvt_convw<<<768, 256, 0, stream>>>(conv0_w, c0sw);
  k_cvt_convw<<<768, 256, 0, stream>>>(conv1_w, c1sw);
  k_cvt_bf16<<<1024, 256, 0, stream>>>(l0_wih, wih0b, 1024L * 512);
  k_cvt_bf16<<<1024, 256, 0, stream>>>(l0_whh, whh0b, 1024L * 256);
  k_cvt_bf16<<<1024, 256, 0, stream>>>(l1_wih, wih1b, 1024L * 256);
  k_cvt_bf16<<<1024, 256, 0, stream>>>(l1_whh, whh1b, 1024L * 256);
  k_uv<<<4, 256, 0, stream>>>(l0_wih, pre_w, pre_b, l0_b, uvec, vvec);

  // parallel front-end (WMMA GEMMs)
  k_conv_ln<<<ROWS / 16, 256, 0, stream>>>(xbf, c0sw, conv0_b, ln0_g, ln0_b, e1bf);
  k_conv_ln<<<ROWS / 16, 256, 0, stream>>>(e1bf, c1sw, conv1_b, ln1_g, ln1_b, e2bf);
  k_gates_pre<<<dim3(ROWS / 16, 8), 256, 0, stream>>>(e2bf, wih0b, target, uvec, vvec, gpre);
  k_out_base<<<ROWS / 8, 256, 0, stream>>>(e2bf, out_w, out_b, out);

  // reset recurrent state + barrier counter (deterministic per launch / graph replay)
  hipMemsetAsync(h0buf, 0, 2 * 64 * 256 * 2, stream);
  hipMemsetAsync(h1buf, 0, 2 * 64 * 256 * 2, stream);
  hipMemsetAsync(bar, 0, 256, stream);

  // sequential scan: 16 persistent WGs, 2 device barriers/step
  k_seq<<<NWG, 256, 0, stream>>>(gpre, whh0b, wih1b, whh1b, l1_b, out_w,
                                 h0buf, h1buf, out, bar);
  k_mask<<<(ROWS + 255) / 256, 256, 0, stream>>>(mask, out, ROWS);
}
